// NLCE_32865089749135
// MI455X (gfx1250) — matrix operand order repaired and checked by reading, not compiled, not run on hardware
//
#include <hip/hip_runtime.h>
#include <hip/hip_bf16.h>

typedef _Float16 v16h __attribute__((ext_vector_type(16)));
typedef _Float16 v8h  __attribute__((ext_vector_type(8)));
typedef float    v8f  __attribute__((ext_vector_type(8)));
typedef unsigned int u32x4 __attribute__((ext_vector_type(4)));
typedef int          i32x4 __attribute__((ext_vector_type(4)));
typedef int          i32x8 __attribute__((ext_vector_type(8)));

constexpr int cB   = 8;
constexpr int cCIN = 512;
constexpr int cC1  = 256;
constexpr int cD   = 128;
constexpr int cK   = 32;       // codewords
constexpr int cN   = 96 * 96;  // 9216

// ---------------------------------------------------------------------------
// TDM: 2D tile (rows x cols f16 elements, row stride ld_elems) -> LDS.
// D# per ISA 8.3/8.4: group0 = {count=1, lds_addr, global_addr, type=2},
// group1 = {data_size=2B, tensor_dim0/1, tile_dim0/1, tensor_dim0_stride}.
// ---------------------------------------------------------------------------
__device__ __forceinline__ void tdm_load_2d(const _Float16* lds_dst,
                                            const _Float16* gsrc,
                                            unsigned rows, unsigned cols,
                                            unsigned ld_elems) {
  unsigned long long ga = (unsigned long long)(uintptr_t)gsrc;
  unsigned lds_off = (unsigned)(uintptr_t)lds_dst;   // low 32 bits = LDS byte offset
  u32x4 g0;
  g0[0] = 1u;                                        // count=1 (valid user D#)
  g0[1] = lds_off;                                   // lds_addr
  g0[2] = (unsigned)ga;                              // global_addr[31:0]
  g0[3] = ((unsigned)(ga >> 32) & 0x01ffffffu) | (2u << 30);  // ga[56:32] | type=2
  i32x8 g1;
  g1[0] = 1 << 16;                                   // data_size=1 -> 2 bytes; no multicast
  g1[1] = (int)((cols & 0xffffu) << 16);             // tensor_dim0 lo16 @ bits63:48
  g1[2] = (int)(((cols >> 16) & 0xffffu) | ((rows & 0xffffu) << 16)); // td0 hi | td1 lo
  g1[3] = (int)(((rows >> 16) & 0xffffu) | ((cols & 0xffffu) << 16)); // td1 hi | tile_dim0
  g1[4] = (int)(rows & 0xffffu);                     // tile_dim1 | tile_dim2=0
  g1[5] = (int)ld_elems;                             // tensor_dim0_stride lo32
  g1[6] = 0;                                         // stride0 hi | stride1 lo
  g1[7] = 0;
  i32x4 z4 = {0, 0, 0, 0};
#if defined(__clang_major__) && (__clang_major__ >= 23)
  i32x8 z8 = {0, 0, 0, 0, 0, 0, 0, 0};
  __builtin_amdgcn_tensor_load_to_lds(g0, g1, z4, z4, z8, 0);
#else
  __builtin_amdgcn_tensor_load_to_lds(g0, g1, z4, z4, 0);
#endif
}

// ---------------------------------------------------------------------------
// Fragment loaders (contiguous k = 16*hi + s order, same for A and B).
// A-frag: lane holds run  [hi*16 .. +15] of its row.
// B-frag: lane holds run  [hi*16 .. +15] of its row.
// ---------------------------------------------------------------------------
__device__ __forceinline__ v16h load_afrag(const _Float16* p) {
  v8h x0 = *(const v8h*)(p);
  v8h x1 = *(const v8h*)(p + 8);
  v16h r;
#pragma unroll
  for (int i = 0; i < 8; ++i) { r[i] = x0[i]; r[i + 8] = x1[i]; }
  return r;
}
__device__ __forceinline__ v16h load_bfrag(const _Float16* p) {
  v8h x0 = *(const v8h*)(p);
  v8h x1 = *(const v8h*)(p + 8);
  v16h r;
#pragma unroll
  for (int i = 0; i < 8; ++i) { r[i] = x0[i]; r[i + 8] = x1[i]; }
  return r;
}

template <int TM>
__device__ __forceinline__ void zero_acc(v8f (&acc)[TM][2]) {
#pragma unroll
  for (int i = 0; i < TM; ++i)
#pragma unroll
    for (int j = 0; j < 2; ++j)
#pragma unroll
      for (int r = 0; r < 8; ++r) acc[i][j][r] = 0.0f;
}

// ---------------------------------------------------------------------------
// TMx2 register-blocked wave GEMM: (TM*16)x32 tile of D = P * Q^T.
// Pa: row-group base (LDS slab or global, stride ldp). Q: n-group base
// (global, stride ldq). Each B-fragment (streamed operand) feeds TM WMMAs.
// ---------------------------------------------------------------------------
template <int TM, bool PFP>
__device__ __forceinline__ void wave_gemm(const _Float16* Pa, long ldp,
                                          const _Float16* __restrict__ Q, long ldq,
                                          int K, int lane, v8f (&acc)[TM][2]) {
  const int mr = lane & 15;
  const int hi = lane >> 4;
  const _Float16* p[TM];
#pragma unroll
  for (int t = 0; t < TM; ++t) p[t] = Pa + (size_t)(mr + 16 * t) * ldp + hi * 16;
  const _Float16* q0 = Q + (size_t)mr * ldq + hi * 16;
  const _Float16* q1 = q0 + (size_t)16 * ldq;
  for (int k0 = 0; k0 < K; k0 += 32) {
    if (k0 + 128 < K) {
      if (PFP) {
#pragma unroll
        for (int t = 0; t < TM; ++t) __builtin_prefetch(p[t] + k0 + 128, 0, 3);
      }
      __builtin_prefetch(q0 + k0 + 128, 0, 3);
      __builtin_prefetch(q1 + k0 + 128, 0, 3);
    }
    v16h B0 = load_bfrag(q0 + k0);
    v16h B1 = load_bfrag(q1 + k0);
#pragma unroll
    for (int t = 0; t < TM; ++t) {
      v16h A = load_afrag(p[t] + k0);
      acc[t][0] = __builtin_amdgcn_wmma_f32_16x16x32_f16(false, A, false, B0, (short)0, acc[t][0], false, false);
      acc[t][1] = __builtin_amdgcn_wmma_f32_16x16x32_f16(false, A, false, B1, (short)0, acc[t][1], false, false);
    }
  }
}

// Block: 128 threads = 4 waves; block tile (TM*16)(M) x 128(N); wave w -> n += w*32.
#define GEMM_PROLOG(TMv)                                        \
  const int lane = threadIdx.x & 31;                            \
  const int wv   = threadIdx.x >> 5;                            \
  const int n0   = blockIdx.x * 128 + wv * 32;                  \
  const int m0   = blockIdx.y * (16 * (TMv));                   \
  const int b    = blockIdx.z;                                  \
  const int mr   = lane & 15;                                   \
  const int hi   = lane >> 4;

// C/D layout: VGPR r -> row m0 + ti*16 + hi*8 + r, col n0 + tj*16 + mr.
#define FOR_TILE(TMv, body)                                     \
  _Pragma("unroll") for (int ti = 0; ti < (TMv); ++ti)          \
  _Pragma("unroll") for (int r = 0; r < 8; ++r) {               \
    const int m = m0 + ti * 16 + hi * 8 + r;                    \
    _Pragma("unroll") for (int tj = 0; tj < 2; ++tj) {          \
      const int n = n0 + tj * 16 + mr;                          \
      body                                                      \
    }                                                           \
  }

// ---------------------------------------------------------------------------
// Small helper kernels
// ---------------------------------------------------------------------------
__global__ void k_cvt_f16(const float* __restrict__ s, _Float16* __restrict__ d, int n) {
  int i = blockIdx.x * blockDim.x + threadIdx.x;
  if (i < n) d[i] = (_Float16)s[i];
}

__global__ void k_cwsq(const float* __restrict__ cw, float* __restrict__ cwsq) {
  int k = threadIdx.x;
  if (k >= cK) return;
  float s = 0.f;
  for (int d = 0; d < cD; ++d) { float v = cw[k * cD + d]; s += v * v; }
  cwsq[k] = s;
}

// (B,C,N) f32  ->  (B,N,C) f16  (LDS-tiled transpose)
__global__ void k_transpose_cvt(const float* __restrict__ src, _Float16* __restrict__ dst,
                                int rows, int cols) {
  __shared__ float tile[32][33];
  const int b = blockIdx.z;
  const float* s = src + (size_t)b * rows * cols;
  _Float16*    d = dst + (size_t)b * rows * cols;
  const int c0 = blockIdx.y * 32, n0 = blockIdx.x * 32;
  const int tx = threadIdx.x, ty = threadIdx.y;  // 32 x 8
  for (int i = 0; i < 32; i += 8)
    tile[ty + i][tx] = s[(size_t)(c0 + ty + i) * cols + (n0 + tx)];
  __syncthreads();
  for (int i = 0; i < 32; i += 8)
    d[(size_t)(n0 + ty + i) * rows + (c0 + tx)] = (_Float16)tile[tx][ty + i];
}

// ---------------------------------------------------------------------------
// Stage A: [Wtheta;Wphi;Wg](768x512) x X^T -> theta/phi (C1,N), g^T (N,C1).
// 64x512 P-slab (64 KB) staged in LDS by the Tensor Data Mover.
// ---------------------------------------------------------------------------
__global__ void __launch_bounds__(128)
k_gemm_tpg(const _Float16* __restrict__ W, const _Float16* __restrict__ XT,
           const float* __restrict__ bias, _Float16* __restrict__ TP,
           _Float16* __restrict__ GT) {
  __shared__ _Float16 slab[64 * cCIN];
  GEMM_PROLOG(4);
  if (threadIdx.x < 32) {
    tdm_load_2d(slab, W + (size_t)m0 * cCIN, 64, cCIN, cCIN);
    __builtin_amdgcn_s_wait_tensorcnt(0);
  }
  __syncthreads();
  const _Float16* Q = XT + (size_t)b * cN * cCIN + (size_t)n0 * cCIN;
  v8f acc[4][2]; zero_acc(acc);
  wave_gemm<4, false>(slab, cCIN, Q, cCIN, cCIN, lane, acc);
  FOR_TILE(4, {
    float v = acc[ti][tj][r] + bias[m];
    if (m < 2 * cC1)
      TP[(size_t)b * (2 * cC1) * cN + (size_t)m * cN + n] = (_Float16)v;
    else
      GT[(size_t)b * cN * cC1 + (size_t)n * cC1 + (m - 2 * cC1)] = (_Float16)v;
  })
}

// Stage B: f = theta * phi^T  (K = 9216, streamed from global)
__global__ void __launch_bounds__(128)
k_gemm_f(const _Float16* __restrict__ TP, float* __restrict__ F) {
  GEMM_PROLOG(4);
  const _Float16* P = TP + (size_t)b * (2 * cC1) * cN + (size_t)m0 * cN;
  const _Float16* Q = TP + (size_t)b * (2 * cC1) * cN + (size_t)cC1 * cN + (size_t)n0 * cN;
  v8f acc[4][2]; zero_acc(acc);
  wave_gemm<4, true>(P, cN, Q, cN, cN, lane, acc);
  FOR_TILE(4, { F[(size_t)b * cC1 * cC1 + (size_t)m * cC1 + n] = acc[ti][tj][r]; })
}

// row softmax (256 cols), f32 -> f16
__global__ void k_softmax_rows(const float* __restrict__ src, _Float16* __restrict__ dst) {
  __shared__ float red[256];
  const int row = blockIdx.x, t = threadIdx.x;
  float v = src[(size_t)row * 256 + t];
  red[t] = v; __syncthreads();
  for (int s = 128; s > 0; s >>= 1) { if (t < s) red[t] = fmaxf(red[t], red[t + s]); __syncthreads(); }
  float m = red[0]; __syncthreads();
  float e = __expf(v - m);
  red[t] = e; __syncthreads();
  for (int s = 128; s > 0; s >>= 1) { if (t < s) red[t] += red[t + s]; __syncthreads(); }
  dst[(size_t)row * 256 + t] = (_Float16)(e / red[0]);
}

// Stage C: y = f * g (stored y^T); 64x256 f-slab (32 KB) via TDM
__global__ void __launch_bounds__(128)
k_gemm_y(const _Float16* __restrict__ FS, const _Float16* __restrict__ GT,
         _Float16* __restrict__ YT) {
  __shared__ _Float16 slab[64 * cC1];
  GEMM_PROLOG(4);
  if (threadIdx.x < 32) {
    tdm_load_2d(slab, FS + (size_t)b * cC1 * cC1 + (size_t)m0 * cC1, 64, cC1, cC1);
    __builtin_amdgcn_s_wait_tensorcnt(0);
  }
  __syncthreads();
  const _Float16* Q = GT + (size_t)b * cN * cC1 + (size_t)n0 * cC1;
  v8f acc[4][2]; zero_acc(acc);
  wave_gemm<4, false>(slab, cC1, Q, cC1, cC1, lane, acc);
  FOR_TILE(4, { YT[(size_t)b * cN * cC1 + (size_t)n * cC1 + m] = (_Float16)acc[ti][tj][r]; })
}

// Stage D: z = W2*y + b2 + X -> d_out (f32) + z^T f16; 64x256 W2-slab via TDM
__global__ void __launch_bounds__(128)
k_gemm_z(const _Float16* __restrict__ W2h, const _Float16* __restrict__ YT,
         const float* __restrict__ b2, const float* __restrict__ X,
         float* __restrict__ Z, _Float16* __restrict__ ZT) {
  __shared__ _Float16 slab[64 * cC1];
  GEMM_PROLOG(4);
  if (threadIdx.x < 32) {
    tdm_load_2d(slab, W2h + (size_t)m0 * cC1, 64, cC1, cC1);
    __builtin_amdgcn_s_wait_tensorcnt(0);
  }
  __syncthreads();
  const _Float16* Q = YT + (size_t)b * cN * cC1 + (size_t)n0 * cC1;
  v8f acc[4][2]; zero_acc(acc);
  wave_gemm<4, false>(slab, cC1, Q, cC1, cC1, lane, acc);
  FOR_TILE(4, {
    size_t zi = (size_t)b * cCIN * cN + (size_t)m * cN + n;
    float v = acc[ti][tj][r] + b2[m] + X[zi];
    Z[zi] = v;
    ZT[(size_t)b * cN * cCIN + (size_t)n * cCIN + m] = (_Float16)v;
  })
}

// Stage E: z' = W3*z -> z'(D,N) + z'^T (N,D); 64x512 W3-slab (64 KB) via TDM
__global__ void __launch_bounds__(128)
k_gemm_zp(const _Float16* __restrict__ W3h, const _Float16* __restrict__ ZT,
          const float* __restrict__ b3, _Float16* __restrict__ ZN,
          _Float16* __restrict__ ZTT) {
  __shared__ _Float16 slab[64 * cCIN];
  GEMM_PROLOG(4);
  if (threadIdx.x < 32) {
    tdm_load_2d(slab, W3h + (size_t)m0 * cCIN, 64, cCIN, cCIN);
    __builtin_amdgcn_s_wait_tensorcnt(0);
  }
  __syncthreads();
  const _Float16* Q = ZT + (size_t)b * cN * cCIN + (size_t)n0 * cCIN;
  v8f acc[4][2]; zero_acc(acc);
  wave_gemm<4, false>(slab, cCIN, Q, cCIN, cCIN, lane, acc);
  FOR_TILE(4, {
    float v = acc[ti][tj][r] + b3[m];
    ZN[(size_t)b * cD * cN + (size_t)m * cN + n]  = (_Float16)v;
    ZTT[(size_t)b * cN * cD + (size_t)n * cD + m] = (_Float16)v;
  })
}

// dot[k,n] = codewords . x_n  (M=32 -> 2x2 blocking); codeword-slab via TDM
__global__ void __launch_bounds__(128)
k_gemm_dot(const _Float16* __restrict__ CWh, const _Float16* __restrict__ ZTT,
           float* __restrict__ DOT) {
  __shared__ _Float16 slab[32 * cD];
  GEMM_PROLOG(2);
  if (threadIdx.x < 32) {
    tdm_load_2d(slab, CWh, 32, cD, cD);
    __builtin_amdgcn_s_wait_tensorcnt(0);
  }
  __syncthreads();
  const _Float16* Q = ZTT + (size_t)b * cN * cD + (size_t)n0 * cD;
  v8f acc[2][2]; zero_acc(acc);
  wave_gemm<2, false>(slab, cD, Q, cD, cD, lane, acc);
  FOR_TILE(2, { DOT[(size_t)b * cK * cN + (size_t)m * cN + n] = acc[ti][tj][r]; })
}

// |x_n|^2 : one wave per n, lane reduction via wave32 shuffles
__global__ void k_xsq(const _Float16* __restrict__ ZTT, float* __restrict__ xsq) {
  const long n = (long)blockIdx.x * (blockDim.x >> 5) + (threadIdx.x >> 5);
  if (n >= (long)cB * cN) return;
  const int lane = threadIdx.x & 31;
  const _Float16* row = ZTT + (size_t)n * cD + lane * 4;
  float s = 0.f;
#pragma unroll
  for (int i = 0; i < 4; ++i) { float x = (float)row[i]; s += x * x; }
  for (int off = 16; off > 0; off >>= 1) s += __shfl_xor(s, off, 32);
  if (lane == 0) xsq[n] = s;
}

// A[n,:] = softmax_k( scale_k * (|x|^2 - 2 dot + |c_k|^2) )
__global__ void k_softmaxA(const float* __restrict__ xsq, const float* __restrict__ DOT,
                           const float* __restrict__ scale, const float* __restrict__ cwsq,
                           _Float16* __restrict__ A) {
  const int i = blockIdx.x * blockDim.x + threadIdx.x;
  if (i >= cB * cN) return;
  const int b = i / cN, n = i - b * cN;
  const float xs = xsq[i];
  const float* dp = DOT + (size_t)b * cK * cN + n;
  float l[cK], mx = -3.4e38f;
#pragma unroll
  for (int k = 0; k < cK; ++k) {
    float v = scale[k] * (xs - 2.0f * dp[(size_t)k * cN] + cwsq[k]);
    l[k] = v; mx = fmaxf(mx, v);
  }
  float s = 0.f;
#pragma unroll
  for (int k = 0; k < cK; ++k) { l[k] = __expf(l[k] - mx); s += l[k]; }
  const float inv = 1.0f / s;
  _Float16* ap = A + (size_t)b * cK * cN + n;
#pragma unroll
  for (int k = 0; k < cK; ++k) ap[(size_t)k * cN] = (_Float16)(l[k] * inv);
}

// sumA[b,k] = sum_n A[b,k,n]  (deterministic block reduction)
__global__ void k_sumA(const _Float16* __restrict__ A, float* __restrict__ sumA) {
  __shared__ float red[256];
  const int bk = blockIdx.x, t = threadIdx.x;
  const _Float16* p = A + (size_t)bk * cN;
  float a = 0.f;
  for (int n = t; n < cN; n += 256) a += (float)p[n];
  red[t] = a; __syncthreads();
  for (int s = 128; s > 0; s >>= 1) { if (t < s) red[t] += red[t + s]; __syncthreads(); }
  if (t == 0) sumA[bk] = red[0];
}

// E[k,d] = sum_n A[k,n] z'[d,n] - sumA[k]*c[k,d]   (K = 9216, M=32, N=128)
__global__ void __launch_bounds__(128)
k_gemm_E(const _Float16* __restrict__ A, const _Float16* __restrict__ ZN,
         const float* __restrict__ sumA, const float* __restrict__ cw,
         float* __restrict__ E) {
  GEMM_PROLOG(2);
  const _Float16* P = A  + (size_t)b * cK * cN;                 // m0 == 0
  const _Float16* Q = ZN + (size_t)b * cD * cN + (size_t)n0 * cN;
  v8f acc[2][2]; zero_acc(acc);
  wave_gemm<2, true>(P, cN, Q, cN, cN, lane, acc);
  FOR_TILE(2, {
    E[(size_t)b * cK * cD + (size_t)m * cD + n] =
        acc[ti][tj][r] - sumA[b * cK + m] * cw[(size_t)m * cD + n];
  })
}

// BN stats per k over (b,d)
__global__ void k_bn_stats(const float* __restrict__ E, float* __restrict__ mean,
                           float* __restrict__ istd) {
  __shared__ float s1[256], s2[256];
  const int k = blockIdx.x, t = threadIdx.x;
  float a = 0.f, q = 0.f;
  for (int i = t; i < cB * cD; i += 256) {
    int bb = i >> 7, d = i & 127;
    float v = E[(size_t)bb * cK * cD + (size_t)k * cD + d];
    a += v; q += v * v;
  }
  s1[t] = a; s2[t] = q; __syncthreads();
  for (int s = 128; s > 0; s >>= 1) {
    if (t < s) { s1[t] += s1[t + s]; s2[t] += s2[t + s]; }
    __syncthreads();
  }
  if (t == 0) {
    float m = s1[0] / (float)(cB * cD);
    float var = s2[0] / (float)(cB * cD) - m * m;
    mean[k] = m;
    istd[k] = rsqrtf(var + 1e-5f);
  }
}

// Es[b,d] = sum_k relu( (E - mean_k) * istd_k )
__global__ void k_Es(const float* __restrict__ E, const float* __restrict__ mean,
                     const float* __restrict__ istd, float* __restrict__ Es) {
  const int i = blockIdx.x * blockDim.x + threadIdx.x;
  if (i >= cB * cD) return;
  const int b = i >> 7, d = i & 127;
  float s = 0.f;
#pragma unroll
  for (int k = 0; k < cK; ++k) {
    float v = (E[(size_t)b * cK * cD + (size_t)k * cD + d] - mean[k]) * istd[k];
    s += fmaxf(v, 0.f);
  }
  Es[i] = s;
}

// gamma[b,c] = sigmoid( Es[b,:] . W_fc[c,:] + b_fc[c] )
__global__ void k_gamma(const float* __restrict__ Es, const float* __restrict__ Wfc,
                        const float* __restrict__ bfc, float* __restrict__ gamma) {
  const int i = blockIdx.x * blockDim.x + threadIdx.x;
  if (i >= cB * cCIN) return;
  const int b = i >> 9, c = i & 511;
  float s = bfc[c];
  for (int d = 0; d < cD; ++d) s += Es[b * cD + d] * Wfc[(size_t)c * cD + d];
  gamma[i] = 1.0f / (1.0f + __expf(-s));
}

// out = z * gamma (in place on d_out)
__global__ void k_scale_out(float* __restrict__ Z, const float* __restrict__ gamma) {
  const size_t idx = (size_t)blockIdx.x * blockDim.x + threadIdx.x;
  const size_t total = (size_t)cB * cCIN * cN;
  if (idx >= total) return;
  const int b = (int)(idx / ((size_t)cCIN * cN));
  const int c = (int)((idx / cN) % cCIN);
  Z[idx] *= gamma[b * cCIN + c];
}

// ---------------------------------------------------------------------------
extern "C" void kernel_launch(void* const* d_in, const int* in_sizes, int n_in,
                              void* d_out, int out_size, void* d_ws, size_t ws_size,
                              hipStream_t stream) {
  (void)in_sizes; (void)n_in; (void)out_size; (void)ws_size;
  const float* X       = (const float*)d_in[0];
  const float* W_theta = (const float*)d_in[1];
  const float* b_theta = (const float*)d_in[2];
  const float* W_phi   = (const float*)d_in[3];
  const float* b_phi   = (const float*)d_in[4];
  const float* W_g     = (const float*)d_in[5];
  const float* b_g     = (const float*)d_in[6];
  const float* W2      = (const float*)d_in[7];
  const float* b2      = (const float*)d_in[8];
  const float* W3      = (const float*)d_in[9];
  const float* b3      = (const float*)d_in[10];
  const float* cw      = (const float*)d_in[11];
  const float* scale   = (const float*)d_in[12];
  const float* W_fc    = (const float*)d_in[13];
  const float* b_fc    = (const float*)d_in[14];
  float* out = (float*)d_out;

  char* ws = (char*)d_ws;
  size_t off = 0;
  auto alloc = [&](size_t bytes) {
    size_t o = off; off += (bytes + 255) & ~(size_t)255; return o;
  };

  // persistent small buffers
  _Float16* wtpg = (_Float16*)(ws + alloc((size_t)768 * cCIN * 2));
  _Float16* w2h  = (_Float16*)(ws + alloc((size_t)cCIN * cC1 * 2));
  _Float16* w3h  = (_Float16*)(ws + alloc((size_t)cD * cCIN * 2));
  _Float16* cwh  = (_Float16*)(ws + alloc((size_t)cK * cD * 2));
  float* cwsq    = (float*)(ws + alloc(cK * 4));
  float* biasT   = (float*)(ws + alloc(768 * 4));
  float* meanb   = (float*)(ws + alloc(cK * 4));
  float* istdb   = (float*)(ws + alloc(cK * 4));
  float* sumA    = (float*)(ws + alloc(cB * cK * 4));
  float* Esb     = (float*)(ws + alloc(cB * cD * 4));
  float* gammab  = (float*)(ws + alloc(cB * cCIN * 4));
  float* Ebuf    = (float*)(ws + alloc((size_t)cB * cK * cD * 4));
  float* F32b    = (float*)(ws + alloc((size_t)cB * cC1 * cC1 * 4));
  _Float16* FS16 = (_Float16*)(ws + alloc((size_t)cB * cC1 * cC1 * 2));
  float* xsq     = (float*)(ws + alloc((size_t)cB * cN * 4));

  // big overlaid regions
  const size_t bigA = (size_t)cB * cN * cCIN * 2;           // 75.5 MB
  char* R_A = ws + alloc(bigA);                             // XT, later ZT
  char* R_B = ws + alloc(bigA);                             // theta/phi, later YT+ZN+ZTT
  char* R_C = ws + alloc((size_t)cB * cN * cC1 * 2);        // GT, later DOT+A16

  _Float16* XT  = (_Float16*)R_A;
  _Float16* ZT  = (_Float16*)R_A;
  _Float16* TP  = (_Float16*)R_B;
  _Float16* YT  = (_Float16*)R_B;
  _Float16* ZN  = (_Float16*)(R_B + (size_t)cB * cN * cC1 * 2);
  _Float16* ZTT = (_Float16*)(R_B + (size_t)cB * cN * cC1 * 2 + (size_t)cB * cN * cD * 2);
  _Float16* GT  = (_Float16*)R_C;
  float*    DOT = (float*)R_C;
  _Float16* A16 = (_Float16*)(R_C + (size_t)cB * cK * cN * 4);

  // ---- weight prep ----
  const int nW1 = cC1 * cCIN;
  k_cvt_f16<<<(nW1 + 255) / 256, 256, 0, stream>>>(W_theta, wtpg, nW1);
  k_cvt_f16<<<(nW1 + 255) / 256, 256, 0, stream>>>(W_phi, wtpg + nW1, nW1);
  k_cvt_f16<<<(nW1 + 255) / 256, 256, 0, stream>>>(W_g, wtpg + 2 * nW1, nW1);
  k_cvt_f16<<<(cCIN * cC1 + 255) / 256, 256, 0, stream>>>(W2, w2h, cCIN * cC1);
  k_cvt_f16<<<(cD * cCIN + 255) / 256, 256, 0, stream>>>(W3, w3h, cD * cCIN);
  k_cvt_f16<<<(cK * cD + 255) / 256, 256, 0, stream>>>(cw, cwh, cK * cD);
  k_cwsq<<<1, 32, 0, stream>>>(cw, cwsq);
  hipMemcpyAsync(biasT,         b_theta, cC1 * 4, hipMemcpyDeviceToDevice, stream);
  hipMemcpyAsync(biasT + cC1,   b_phi,   cC1 * 4, hipMemcpyDeviceToDevice, stream);
  hipMemcpyAsync(biasT + 2*cC1, b_g,     cC1 * 4, hipMemcpyDeviceToDevice, stream);

  // ---- X transpose+convert ----
  k_transpose_cvt<<<dim3(cN / 32, cCIN / 32, cB), dim3(32, 8), 0, stream>>>(X, XT, cCIN, cN);

  // ---- pipeline ----
  k_gemm_tpg<<<dim3(cN / 128, 768 / 64, cB), 128, 0, stream>>>(wtpg, XT, biasT, TP, GT);
  k_gemm_f<<<dim3(cC1 / 128, cC1 / 64, cB), 128, 0, stream>>>(TP, F32b);
  k_softmax_rows<<<cB * cC1, 256, 0, stream>>>(F32b, FS16);
  k_gemm_y<<<dim3(cN / 128, cC1 / 64, cB), 128, 0, stream>>>(FS16, GT, YT);
  k_gemm_z<<<dim3(cN / 128, cCIN / 64, cB), 128, 0, stream>>>(w2h, YT, b2, X, out, ZT);
  k_gemm_zp<<<dim3(cN / 128, cD / 64, cB), 128, 0, stream>>>(w3h, ZT, b3, ZN, ZTT);
  k_gemm_dot<<<dim3(cN / 128, 1, cB), 128, 0, stream>>>(cwh, ZTT, DOT);
  k_xsq<<<(cB * cN) / 4, 128, 0, stream>>>(ZTT, xsq);
  k_softmaxA<<<(cB * cN + 255) / 256, 256, 0, stream>>>(xsq, DOT, scale, cwsq, A16);
  k_sumA<<<cB * cK, 256, 0, stream>>>(A16, sumA);
  k_gemm_E<<<dim3(1, 1, cB), 128, 0, stream>>>(A16, ZN, sumA, cw, Ebuf);
  k_bn_stats<<<cK, 256, 0, stream>>>(Ebuf, meanb, istdb);
  k_Es<<<(cB * cD + 255) / 256, 256, 0, stream>>>(Ebuf, meanb, istdb, Esb);
  k_gamma<<<(cB * cCIN + 255) / 256, 256, 0, stream>>>(Esb, W_fc, b_fc, gammab);

  const size_t total = (size_t)cB * cCIN * cN;
  k_scale_out<<<(unsigned)((total + 255) / 256), 256, 0, stream>>>(out, gammab);
}
